// BKNOBlock_71674414235846
// MI455X (gfx1250) — matrix operand order repaired and checked by reading, not compiled, not run on hardware
//
#include <hip/hip_runtime.h>

typedef __attribute__((ext_vector_type(16))) _Float16 v16h;
typedef __attribute__((ext_vector_type(8)))  _Float16 v8h;
typedef __attribute__((ext_vector_type(8)))  float    v8f;

#define B_  2
#define C_  32
#define D_  32
#define H_  64
#define W_  64
#define HW_   (H_ * W_)           // 4096
#define DHW_  (D_ * HW_)          // 131072
#define NTAP  27
#define WQ_ELEMS (NTAP * C_ * C_) // 27648
#define ROWS_PER_BLK 4            // output h-rows per workgroup
#define STAGE_ROWS   (ROWS_PER_BLK + 2)       // 6 (halo h-1 .. h+4)
#define XL_ROW_HALFS (W_ * C_)                // 2048 halfs = 4 KB per staged row
#define XL_HALFS     (3 * STAGE_ROWS * XL_ROW_HALFS) // 36864 halfs = 72 KB

union AU { v16h v; v8h h[2]; };

__device__ __forceinline__ float softplusf(float x) { return log1pf(expf(x)); }

// ---------------------------------------------------------------------------
// Kernel 0: wq[tap][co][ci] = sum_j softplus(lambda_j) * (logits[j][co][ci][tap] >= 0)
// ---------------------------------------------------------------------------
__global__ __launch_bounds__(256) void prep_w(const float* __restrict__ logits,
                                              const float* __restrict__ lamb_raw,
                                              _Float16* __restrict__ wq) {
    int idx = blockIdx.x * 256 + threadIdx.x;
    if (idx >= WQ_ELEMS) return;
    int ci  = idx & 31;
    int co  = (idx >> 5) & 31;
    int tap = idx >> 10;
    float acc = 0.0f;
#pragma unroll
    for (int j = 0; j < 4; ++j) {
        float lj = softplusf(lamb_raw[j]);
        float lg = logits[(size_t)j * (C_ * C_ * NTAP) + ((size_t)co * C_ + ci) * NTAP + tap];
        acc += (lg >= 0.0f) ? lj : 0.0f;
    }
    wq[((size_t)tap * C_ + co) * C_ + ci] = (_Float16)acc;
}

// ---------------------------------------------------------------------------
// Kernel 1: x[b][d][h][w][c] (f16, channel-last) = sum_r softplus(beta_r)*(a >= thr_r)
// One block per (b,d,h) row: coalesced a read, LDS transpose, coalesced 16B writes.
// ---------------------------------------------------------------------------
__global__ __launch_bounds__(256) void prep_x(const float* __restrict__ a,
                                              const float* __restrict__ thr,
                                              const float* __restrict__ beta_raw,
                                              _Float16* __restrict__ xq) {
    __shared__ _Float16 tile[W_ * C_];   // [w][c], 4 KB
    int blk = blockIdx.x;                // b*D*H + d*H + h
    int h = blk & 63;
    int d = (blk >> 6) & 31;
    int b = blk >> 11;
    int t = threadIdx.x;

    float b0 = softplusf(beta_raw[0]);
    float b1 = softplusf(beta_raw[1]);
    float t0 = thr[0], t1 = thr[1];

    size_t abase = (size_t)b * (C_ * DHW_) + (size_t)d * HW_ + (size_t)h * W_;
#pragma unroll
    for (int j = 0; j < 8; ++j) {
        int e = t + 256 * j;          // 0..2047
        int c = e >> 6;
        int w = e & 63;               // coalesced over threads
        float av = a[abase + (size_t)c * DHW_ + w];
        float v  = (av >= t0 ? b0 : 0.0f) + (av >= t1 ? b1 : 0.0f);
        tile[w * C_ + c] = (_Float16)v;
    }
    __syncthreads();
    v8h*       dst = (v8h*)(xq + (size_t)blk * XL_ROW_HALFS);
    const v8h* src = (const v8h*)tile;
    dst[t] = src[t];
}

// ---------------------------------------------------------------------------
// Kernel 2: LDS-staged implicit-GEMM conv via v_wmma_f32_16x16x32_f16.
// Block = 8 waves, owns (b, d, h0..h0+3, w 0..63, all 32 c_out).
// Stages 3 d-planes x 6 h-rows of channel-last x into LDS (zero-filled halo),
// so all 27 taps read B from LDS; only the w=+-1 edge needs a per-lane select.
// Wave: (h-row = wv>>1, w-half = wv&1) -> 2 spatial N-tiles x 2 c_out tiles
// = 4 accumulators; one A-pair + 2 B vectors feed 4 WMMAs per tap (108 total).
// ---------------------------------------------------------------------------
__global__ __launch_bounds__(256) void conv_wmma(const _Float16* __restrict__ xq,
                                                 const _Float16* __restrict__ wq,
                                                 const float* __restrict__ a,
                                                 const float* __restrict__ omega_p,
                                                 float* __restrict__ out) {
    __shared__ _Float16 wlds[WQ_ELEMS];  // 55296 B
    __shared__ _Float16 xlds[XL_HALFS];  // 73728 B : [plane 0..2][row 0..5][w][ci]
    int tid = threadIdx.x;

    int hblk = blockIdx.x & 15;
    int d    = (blockIdx.x >> 4) & 31;
    int b    = blockIdx.x >> 9;
    int h0   = hblk * ROWS_PER_BLK;

    // ---- stage weights into LDS (16B chunks) ----
    {
        const v8h* src = (const v8h*)wq;
        v8h*       dst = (v8h*)wlds;
        for (int i = tid; i < WQ_ELEMS / 8; i += 256) dst[i] = src[i];
    }

    // ---- stage x block with halo: 18 (plane,row) pairs x 256 chunks of 16B ----
    {
        v8h zero8 = {};
        v8h* dst = (v8h*)xlds;
        const v8h* src = (const v8h*)xq;
#pragma unroll
        for (int pr = 0; pr < 3 * STAGE_ROWS; ++pr) {
            int plane = pr / STAGE_ROWS;           // folds at compile time
            int row   = pr - plane * STAGE_ROWS;
            int d2 = d + plane - 1;
            int h2 = h0 + row - 1;
            bool valid = ((unsigned)d2 < (unsigned)D_) && ((unsigned)h2 < (unsigned)H_);
            size_t srow = (((size_t)b * D_ + (size_t)(valid ? d2 : 0)) * H_ +
                           (size_t)(valid ? h2 : 0));
            v8h vvec = src[srow * (XL_ROW_HALFS / 8) + tid];
            dst[pr * (XL_ROW_HALFS / 8) + tid] = valid ? vvec : zero8;
        }
    }
    __syncthreads();

    int lane  = tid & 31;
    int m     = lane & 15;      // N column / A row sub-index
    int hi    = lane >> 4;      // K-half select: contiguous k = hi*16 + s in A and B
    int wv    = tid >> 5;       // wave in block: 0..7
    int wr    = wv >> 1;        // output row 0..3
    int wbase = (wv & 1) * 32;  // w half: 0 or 32

    size_t pbase = (size_t)d * HW_ + (size_t)(h0 + wr) * W_ + wbase + m;
    size_t obase = (size_t)b * (C_ * DHW_) + pbase;

    // warm the epilogue 'a' lines while the MAC loop runs
    __builtin_prefetch(a + obase, 0, 3);
    __builtin_prefetch(a + obase + (size_t)16 * DHW_, 0, 3);

    v8f acc00 = {};  // ntile0, c_out 0..15
    v8f acc01 = {};  // ntile0, c_out 16..31
    v8f acc10 = {};  // ntile1, c_out 0..15
    v8f acc11 = {};  // ntile1, c_out 16..31
    v16h zerov = {};

#pragma unroll
    for (int kd = 0; kd < 3; ++kd) {
#pragma unroll
        for (int kh = 0; kh < 3; ++kh) {
            int base = (kd * STAGE_ROWS + wr + kh) * XL_ROW_HALFS;
#pragma unroll
            for (int kw = 0; kw < 3; ++kw) {
                int tap = (kd * 3 + kh) * 3 + kw;

                // B operands from LDS: 32 contiguous f16 of one staged row;
                // lanes 0-15: ci 0..15, lanes 16-31: ci 16..31.
                int w2a = wbase + m + kw - 1;           // ntile 0 column
                int w2b = w2a + 16;                     // ntile 1 column
                bool va = (unsigned)w2a < (unsigned)W_;
                bool vb = (unsigned)w2b < (unsigned)W_;
                v16h bxa = *(const v16h*)(xlds + base + (va ? w2a * C_ : 0) + hi * 16);
                v16h bxb = *(const v16h*)(xlds + base + (vb ? w2b * C_ : 0) + hi * 16);
                bxa = va ? bxa : zerov;
                bxb = vb ? bxb : zerov;

                // A operands: row co = m (+16), K contiguous {hi*16..hi*16+15}
                int arow = (tap * C_ + m) * C_ + hi * 16;
                AU ua, ub;
                ua.h[0] = *(const v8h*)(wlds + arow);
                ua.h[1] = *(const v8h*)(wlds + arow + 8);
                ub.h[0] = *(const v8h*)(wlds + arow + 16 * C_);
                ub.h[1] = *(const v8h*)(wlds + arow + 16 * C_ + 8);

                acc00 = __builtin_amdgcn_wmma_f32_16x16x32_f16(
                    false, ua.v, false, bxa, (short)0, acc00, false, false);
                acc01 = __builtin_amdgcn_wmma_f32_16x16x32_f16(
                    false, ub.v, false, bxa, (short)0, acc01, false, false);
                acc10 = __builtin_amdgcn_wmma_f32_16x16x32_f16(
                    false, ua.v, false, bxb, (short)0, acc10, false, false);
                acc11 = __builtin_amdgcn_wmma_f32_16x16x32_f16(
                    false, ub.v, false, bxb, (short)0, acc11, false, false);
            }
        }
    }

    float omega = omega_p[0];
    // D layout: VGPR r -> M(co) = r + hi*8 ; lane&15 -> N(spatial) -> coalesced stores
#pragma unroll
    for (int r = 0; r < 8; ++r) {
        int co0 = hi * 8 + r;
        int co1 = co0 + 16;
        {
            size_t idx = obase + (size_t)co0 * DHW_;
            float z = acc00[r] + omega * a[idx];
            out[idx] = 0.5f * z * (1.0f + erff(z * 0.70710678118654752f));
        }
        {
            size_t idx = obase + (size_t)co1 * DHW_;
            float z = acc01[r] + omega * a[idx];
            out[idx] = 0.5f * z * (1.0f + erff(z * 0.70710678118654752f));
        }
        {
            size_t idx = obase + 16 + (size_t)co0 * DHW_;
            float z = acc10[r] + omega * a[idx];
            out[idx] = 0.5f * z * (1.0f + erff(z * 0.70710678118654752f));
        }
        {
            size_t idx = obase + 16 + (size_t)co1 * DHW_;
            float z = acc11[r] + omega * a[idx];
            out[idx] = 0.5f * z * (1.0f + erff(z * 0.70710678118654752f));
        }
    }
}

// ---------------------------------------------------------------------------
// Launch
// ---------------------------------------------------------------------------
extern "C" void kernel_launch(void* const* d_in, const int* in_sizes, int n_in,
                              void* d_out, int out_size, void* d_ws, size_t ws_size,
                              hipStream_t stream) {
    const float* a         = (const float*)d_in[0];  // [2,32,32,64,64]
    const float* thr       = (const float*)d_in[1];  // [2]
    const float* beta_raw  = (const float*)d_in[2];  // [2]
    const float* logits    = (const float*)d_in[3];  // [4,32,32,3,3,3]
    const float* lamb_raw  = (const float*)d_in[4];  // [4]
    const float* omega     = (const float*)d_in[5];  // [1]
    float* out = (float*)d_out;

    // scratch layout: [0, 16MB) = channel-last f16 x ; then f16 weights (55 KB)
    _Float16* xq  = (_Float16*)d_ws;
    _Float16* wqp = (_Float16*)((char*)d_ws + (size_t)16 * 1024 * 1024);

    prep_w<<<(WQ_ELEMS + 255) / 256, 256, 0, stream>>>(logits, lamb_raw, wqp);
    prep_x<<<B_ * D_ * H_, 256, 0, stream>>>(a, thr, beta_raw, xq);
    conv_wmma<<<B_ * D_ * (H_ / ROWS_PER_BLK), 256, 0, stream>>>(xq, wqp, a, omega, out);
}
